// DomainMoE_25950192402966
// MI455X (gfx1250) — matrix operand order repaired and checked by reading, not compiled, not run on hardware
//
#include <hip/hip_runtime.h>
#include <cstdint>

typedef __attribute__((ext_vector_type(16))) _Float16 v16h;
typedef __attribute__((ext_vector_type(8)))  float    v8f;
typedef unsigned int u32;
typedef __attribute__((ext_vector_type(4))) u32 u32x4;
typedef __attribute__((ext_vector_type(8))) int i32x8;
typedef __attribute__((ext_vector_type(4))) int i32x4;

#define NEXP 6

#if defined(__gfx1250__) && defined(__has_builtin)
#if __has_builtin(__builtin_amdgcn_tensor_load_to_lds) && \
    __has_builtin(__builtin_amdgcn_s_wait_tensorcnt)
#define USE_TDM 1
#endif
#endif
#ifndef USE_TDM
#define USE_TDM 0
#endif

union FragH {
    v16h v;
    u32  u[8];
};

__device__ __forceinline__ u32 pack2h(float a, float b) {
    union { u32 u; _Float16 h[2]; } t;
    t.h[0] = (_Float16)a;
    t.h[1] = (_Float16)b;
    return t.u;
}

__device__ __forceinline__ float gelu_exact(float x) {
    return 0.5f * x * (1.0f + erff(x * 0.70710678118654752f));
}

// Low 32 bits of a generic pointer to a __shared__ object == LDS byte offset
// (flat LDS aperture maps addr[31:0] -> LDS address, ISA 10.2).
__device__ __forceinline__ u32 lds_offset(const void* p) {
    return (u32)(uintptr_t)p;
}

#if USE_TDM
// ---------------------------------------------------------------------------
// TDM: DMA a 128-row x 32-halfword tile (rows K-contiguous, row stride =
// Ktotal halves) from global f16 into LDS with 1-DWORD padding every 16
// DWORDs (=> 17-uint LDS pitch). D# per ISA 8.3/8.4; groups 2/3 zero (2D).
// 6-arg builtin form (amdgpu-toolchain / clang-23).
// ---------------------------------------------------------------------------
__device__ __forceinline__ void tdm_load_tile(const _Float16* gsrc, u32 ldsOff,
                                              int Ktotal, int Rtotal) {
    const unsigned long long ga = (unsigned long long)(uintptr_t)gsrc;
    u32x4 g0;
    g0[0] = 1u;                                            // count=1, no gather
    g0[1] = ldsOff;                                        // lds_addr (bytes)
    g0[2] = (u32)ga;                                       // global_addr[31:0]
    g0[3] = (u32)((ga >> 32) & 0x01FFFFFFu) | (2u << 30);  // addr[56:32] | type=2
    i32x8 g1;
    g1[0] = (int)((1u << 16)      // data_size = 2 bytes
                | (1u << 20)      // pad_enable
                | (3u << 22));    // pad_interval: 16 DWORDs; pad_amount: 1 DWORD
    g1[1] = (int)(((u32)Ktotal & 0xFFFFu) << 16);                         // dim0 lo16
    g1[2] = (int)(((u32)Ktotal >> 16) | (((u32)Rtotal & 0xFFFFu) << 16)); // dim0 hi | dim1 lo
    g1[3] = (int)(((u32)Rtotal >> 16) | (32u << 16));                     // dim1 hi | tile_dim0=32
    g1[4] = 128;                                                          // tile_dim1=128, tile_dim2=0
    g1[5] = Ktotal;                                                       // tensor_dim0_stride lo32
    g1[6] = 0;                                                            // stride hi16 | dim1_stride lo16
    g1[7] = 0;
    const i32x4 gz4 = {0, 0, 0, 0};
    const i32x8 gz8 = {0, 0, 0, 0, 0, 0, 0, 0};
    __builtin_amdgcn_tensor_load_to_lds(g0, g1, gz4, gz4, gz8, 0);
}
#endif

// ---------------------------------------------------------------------------
// Router: one wave32 per token; shuffle-reduced dot products, lane 0 does
// softmax / top-2 / normalized combine weights + aux-loss accumulation.
// ---------------------------------------------------------------------------
__global__ void moe_router_kernel(const float* __restrict__ x,
                                  const float* __restrict__ wr,
                                  float* __restrict__ comb,
                                  float* __restrict__ probsSum,
                                  int N, int D) {
    const int lane  = threadIdx.x & 31;
    const int wid   = threadIdx.x >> 5;
    const int token = blockIdx.x * 8 + wid;
    if (token >= N) return;

    float p[NEXP] = {0.f, 0.f, 0.f, 0.f, 0.f, 0.f};
    const float* xr = x + (size_t)token * D;
    for (int d = lane; d < D; d += 32) {
        const float xv = xr[d];
#pragma unroll
        for (int e = 0; e < NEXP; ++e) p[e] += xv * wr[d * NEXP + e];
    }
#pragma unroll
    for (int off = 16; off > 0; off >>= 1) {
#pragma unroll
        for (int e = 0; e < NEXP; ++e) p[e] += __shfl_xor(p[e], off, 32);
    }

    if (lane == 0) {
        float mx = p[0];
#pragma unroll
        for (int e = 1; e < NEXP; ++e) mx = fmaxf(mx, p[e]);
        float pr[NEXP], s = 0.f;
#pragma unroll
        for (int e = 0; e < NEXP; ++e) { pr[e] = expf(p[e] - mx); s += pr[e]; }
        const float inv = 1.0f / s;
#pragma unroll
        for (int e = 0; e < NEXP; ++e) pr[e] *= inv;

        int i1 = 0;
#pragma unroll
        for (int e = 1; e < NEXP; ++e) if (pr[e] > pr[i1]) i1 = e;
        int i2 = (i1 == 0) ? 1 : 0;
#pragma unroll
        for (int e = 0; e < NEXP; ++e) if (e != i1 && pr[e] > pr[i2]) i2 = e;
        const float wsum = pr[i1] + pr[i2];

        float c[NEXP] = {0.f, 0.f, 0.f, 0.f, 0.f, 0.f};
        c[i1] = pr[i1] / wsum;
        c[i2] = pr[i2] / wsum;
#pragma unroll
        for (int e = 0; e < NEXP; ++e) comb[(size_t)token * NEXP + e] = c[e];
#pragma unroll
        for (int e = 0; e < NEXP; ++e) atomicAdd(&probsSum[e], pr[e]);
    }
}

// ---------------------------------------------------------------------------
// f32 -> f16 elementwise (n multiple of 4).
// ---------------------------------------------------------------------------
__global__ void moe_cvt_f16_kernel(const float* __restrict__ src,
                                   _Float16* __restrict__ dst, size_t n4) {
    const size_t i = (size_t)blockIdx.x * blockDim.x + threadIdx.x;
    if (i < n4) {
        const float4 v = *(const float4*)(src + i * 4);
        u32* d = (u32*)(dst + i * 4);
        d[0] = pack2h(v.x, v.y);
        d[1] = pack2h(v.z, v.w);
    }
}

// ---------------------------------------------------------------------------
// f32 (K x N, row-major) -> f16 transposed (N x K). 32x32 LDS tile, 32x8 block.
// ---------------------------------------------------------------------------
__global__ void moe_cvt_tr_kernel(const float* __restrict__ src,
                                  _Float16* __restrict__ dst, int K, int N) {
    __shared__ float tile[32][33];
    const int n0 = blockIdx.x * 32, k0 = blockIdx.y * 32;
    const int tx = threadIdx.x, ty = threadIdx.y;
#pragma unroll
    for (int r = 0; r < 32; r += 8)
        tile[ty + r][tx] = src[(size_t)(k0 + ty + r) * N + n0 + tx];
    __syncthreads();
#pragma unroll
    for (int r = 0; r < 32; r += 8)
        dst[(size_t)(n0 + ty + r) * K + k0 + tx] = (_Float16)tile[tx][ty + r];
}

// ---------------------------------------------------------------------------
// WMMA GEMM: C(M x N) = A(M x K, f16, ld=K) @ Bt(N x K, f16, ld=K)^T.
// Block = 8 waves, 128x128 tile; wave = 32x64 = 2x4 WMMA accumulators.
// K staged in 32-wide slabs: TDM double-buffered (wave 0 issues, TENSORcnt
// gates) or plain uint4 fallback. LDS pitch = 17 uints (TDM pad feature).
// EPILOGUE 0: gelu -> f16 H1out.  EPILOGUE 1: comb-scaled f32 (+)= Out.
// ---------------------------------------------------------------------------
template <int EPILOGUE>
__global__ __launch_bounds__(256)
void moe_gemm_wmma(const _Float16* __restrict__ A,
                   const _Float16* __restrict__ Bt,
                   _Float16* __restrict__ H1out,
                   float* __restrict__ Out,
                   const float* __restrict__ combp,
                   int Mdim, int Ndim, int Kdim, int expert, int accum) {
    __shared__ u32 lds_a[2][128][17];
    __shared__ u32 lds_b[2][128][17];

    const int tid     = threadIdx.x;
    const int lane    = tid & 31;
    const int wid     = tid >> 5;
    const int lane15  = lane & 15;
    const int waveRow = (wid & 3) * 32;
    const int waveCol = (wid >> 2) * 64;
    const int rowBase = blockIdx.y * 128;
    const int colBase = blockIdx.x * 128;

    const v8f zacc = {0.f, 0.f, 0.f, 0.f, 0.f, 0.f, 0.f, 0.f};
    v8f acc[2][4];
#pragma unroll
    for (int tm = 0; tm < 2; ++tm)
#pragma unroll
        for (int tn = 0; tn < 4; ++tn) acc[tm][tn] = zacc;

    // Fragment K-pair bases (ISA 7.12.2): A 16x32 lanes>=16 -> K base 8;
    // B 32x16 lanes>=16 -> K base 8 (same K order as A).
    const int aKb = (lane < 16) ? 0 : 4;
    const int bKb = (lane < 16) ? 0 : 4;

    const int nk = Kdim / 32;
#if USE_TDM
    const _Float16* aBase = A + (size_t)rowBase * Kdim;
    const _Float16* bBase = Bt + (size_t)colBase * Kdim;
    if (wid == 0) {
        tdm_load_tile(aBase, lds_offset(&lds_a[0][0][0]), Kdim, Mdim);
        tdm_load_tile(bBase, lds_offset(&lds_b[0][0][0]), Kdim, Ndim);
    }
#endif

    for (int i = 0; i < nk; ++i) {
#if USE_TDM
        const int buf = i & 1;
        if (wid == 0) {
            if (i + 1 < nk) {
                tdm_load_tile(aBase + (size_t)(i + 1) * 32,
                              lds_offset(&lds_a[(i + 1) & 1][0][0]), Kdim, Mdim);
                tdm_load_tile(bBase + (size_t)(i + 1) * 32,
                              lds_offset(&lds_b[(i + 1) & 1][0][0]), Kdim, Ndim);
                __builtin_amdgcn_s_wait_tensorcnt(2);  // tile i complete
            } else {
                __builtin_amdgcn_s_wait_tensorcnt(0);
            }
        }
#else
        const int buf = 0;
        // Fallback staging: A and Bt rows are K-contiguous f16; uint4 copies.
#pragma unroll
        for (int t = 0; t < 2; ++t) {
            const int idx = t * 256 + tid;   // 0..511
            const int m   = idx >> 2;
            const int kq  = idx & 3;         // 8 halves each
            const uint4 av = *(const uint4*)(A + (size_t)(rowBase + m) * Kdim + i * 32 + kq * 8);
            lds_a[0][m][kq * 4 + 0] = av.x;
            lds_a[0][m][kq * 4 + 1] = av.y;
            lds_a[0][m][kq * 4 + 2] = av.z;
            lds_a[0][m][kq * 4 + 3] = av.w;
            const uint4 bv = *(const uint4*)(Bt + (size_t)(colBase + m) * Kdim + i * 32 + kq * 8);
            lds_b[0][m][kq * 4 + 0] = bv.x;
            lds_b[0][m][kq * 4 + 1] = bv.y;
            lds_b[0][m][kq * 4 + 2] = bv.z;
            lds_b[0][m][kq * 4 + 3] = bv.w;
        }
#endif
        __syncthreads();

        const u32 (*la)[17] = lds_a[buf];
        const u32 (*lb)[17] = lds_b[buf];
        FragH fa0, fa1;
#pragma unroll
        for (int j = 0; j < 8; ++j) {
            const int kk = aKb + (j & 3) + ((j >> 2) << 3);
            fa0.u[j] = la[waveRow + lane15][kk];
            fa1.u[j] = la[waveRow + 16 + lane15][kk];
        }
#pragma unroll
        for (int tn = 0; tn < 4; ++tn) {
            FragH fb;
#pragma unroll
            for (int j = 0; j < 8; ++j)
                fb.u[j] = lb[waveCol + tn * 16 + lane15][bKb + (j & 3) + ((j >> 2) << 3)];
            acc[0][tn] = __builtin_amdgcn_wmma_f32_16x16x32_f16(
                false, fa0.v, false, fb.v, (short)0, acc[0][tn], false, false);
            acc[1][tn] = __builtin_amdgcn_wmma_f32_16x16x32_f16(
                false, fa1.v, false, fb.v, (short)0, acc[1][tn], false, false);
        }
        __syncthreads();
    }

    // Epilogue: C/D layout -> col = lane%16, row = vgpr + 8*(lane/16).
    const int rOff = (lane >> 4) << 3;
#pragma unroll
    for (int tm = 0; tm < 2; ++tm) {
#pragma unroll
        for (int tn = 0; tn < 4; ++tn) {
#pragma unroll
            for (int v = 0; v < 8; ++v) {
                const int r = rowBase + waveRow + tm * 16 + v + rOff;
                const int c = colBase + waveCol + tn * 16 + lane15;
                const float a = acc[tm][tn][v];
                if (EPILOGUE == 0) {
                    H1out[(size_t)r * Ndim + c] = (_Float16)gelu_exact(a);
                } else {
                    const float cw  = combp[(size_t)r * NEXP + expert];
                    const float val = cw * a;
                    const size_t o  = (size_t)r * Ndim + c;
                    if (accum) Out[o] += val;
                    else       Out[o]  = val;
                }
            }
        }
    }
    (void)Mdim;
}

__global__ void moe_zero6_kernel(float* p) {
    if (threadIdx.x < NEXP && blockIdx.x == 0) p[threadIdx.x] = 0.f;
}

__global__ void moe_aux_kernel(const float* __restrict__ probsSum,
                               float* __restrict__ outAux, float invN) {
    if (threadIdx.x == 0 && blockIdx.x == 0) {
        float s = 0.f;
#pragma unroll
        for (int e = 0; e < NEXP; ++e) {
            const float d = probsSum[e] * invN - (1.0f / NEXP);
            s += d * d;
        }
        outAux[0] = 0.01f * (s / NEXP);
    }
}

// ---------------------------------------------------------------------------
// Inputs: x (B,T,D) f32; w_router (D,6) f32; w1 (6,D,H) f32; w2 (6,H,D) f32.
// Output: out (B,T,D) f32 ++ aux_loss (1 f32).
// ---------------------------------------------------------------------------
extern "C" void kernel_launch(void* const* d_in, const int* in_sizes, int n_in,
                              void* d_out, int out_size, void* d_ws, size_t ws_size,
                              hipStream_t stream) {
    const float* x  = (const float*)d_in[0];
    const float* wr = (const float*)d_in[1];
    const float* w1 = (const float*)d_in[2];
    const float* w2 = (const float*)d_in[3];
    float* out = (float*)d_out;

    const int D = in_sizes[1] / NEXP;        // 1024
    const int N = in_sizes[0] / D;           // 8192 tokens
    const int H = in_sizes[2] / (NEXP * D);  // 4096

    // ws: comb (N*6 f32) | probsSum | pad | xh (N*D f16) | h1 (N*H f16) |
    //     w1t (H*D f16, reused per expert) | w2t (D*H f16, reused per expert)
    float* comb     = (float*)d_ws;
    float* probsSum = comb + (size_t)N * NEXP;
    size_t off = (((size_t)(N * NEXP + 8) * sizeof(float)) + 255) & ~(size_t)255;
    _Float16* xh  = (_Float16*)((char*)d_ws + off); off += (size_t)N * D * sizeof(_Float16);
    _Float16* h1  = (_Float16*)((char*)d_ws + off); off += (size_t)N * H * sizeof(_Float16);
    _Float16* w1t = (_Float16*)((char*)d_ws + off); off += (size_t)H * D * sizeof(_Float16);
    _Float16* w2t = (_Float16*)((char*)d_ws + off);

    moe_zero6_kernel<<<1, 32, 0, stream>>>(probsSum);
    moe_router_kernel<<<N / 8, 256, 0, stream>>>(x, wr, comb, probsSum, N, D);
    moe_cvt_f16_kernel<<<(unsigned)(((size_t)N * D / 4 + 255) / 256), 256, 0, stream>>>(
        x, xh, (size_t)N * D / 4);

    for (int e = 0; e < NEXP; ++e) {
        // w1[e] (D x H) -> w1t (H x D, f16, K-contiguous)
        moe_cvt_tr_kernel<<<dim3(H / 32, D / 32), dim3(32, 8), 0, stream>>>(
            w1 + (size_t)e * D * H, w1t, D, H);
        // h = gelu(x @ w1[e]) : M=N, N=H, K=D
        moe_gemm_wmma<0><<<dim3(H / 128, N / 128), 256, 0, stream>>>(
            xh, w1t, h1, nullptr, nullptr, N, H, D, e, 0);
        // w2[e] (H x D) -> w2t (D x H, f16)
        moe_cvt_tr_kernel<<<dim3(D / 32, H / 32), dim3(32, 8), 0, stream>>>(
            w2 + (size_t)e * H * D, w2t, H, D);
        // out (+)= comb[:,e] * (h @ w2[e]) : M=N, N=D, K=H
        moe_gemm_wmma<1><<<dim3(D / 128, N / 128), 256, 0, stream>>>(
            h1, w2t, nullptr, out, comb, N, D, H, e, e > 0);
    }

    moe_aux_kernel<<<1, 1, 0, stream>>>(probsSum, out + (size_t)N * D, 1.0f / (float)N);
}
